// GroupedQueryAttention_49460843380757
// MI455X (gfx1250) — matrix operand order repaired and checked by reading, not compiled, not run on hardware
//
#include <hip/hip_runtime.h>

// ---------------------------------------------------------------------------
// GQA for MI455X (gfx1250): bf16 WMMA everywhere (projections + attention),
// async global->LDS DMA for the V tiles, prefetch for K/V streams.
// ---------------------------------------------------------------------------

typedef __bf16 bf16;
typedef __attribute__((ext_vector_type(8)))  bf16  v8bf;
typedef __attribute__((ext_vector_type(16))) bf16  v16bf;
typedef __attribute__((ext_vector_type(8)))  float v8f;

#define DEVINL __device__ __forceinline__

constexpr int BATCH = 2;
constexpr int T     = 2048;
constexpr int E     = 2048;
constexpr int G     = 4;          // kv groups
constexpr int R     = 4;          // heads per group
constexpr int HD    = 128;        // head dim
constexpr int KV    = G * HD;     // 512
constexpr int M     = BATCH * T;  // 4096 rows

DEVINL v16bf cat16(v8bf lo, v8bf hi) {
  return __builtin_shufflevector(lo, hi, 0,1,2,3,4,5,6,7,8,9,10,11,12,13,14,15);
}

DEVINL v8f wmma_bf16(v16bf a, v16bf b, v8f c) {
  return __builtin_amdgcn_wmma_f32_16x16x32_bf16(false, a, false, b, (short)0, c,
                                                 false, false);
}

// 16-byte async copy global -> LDS, tracked by ASYNCcnt.
DEVINL void async_copy_b128(const bf16* gsrc, bf16* lds_dst) {
  unsigned ldsoff = (unsigned)(uintptr_t)lds_dst;   // AS3 offset lives in low bits
  asm volatile("global_load_async_to_lds_b128 %0, %1, off"
               :: "v"(ldsoff), "v"(gsrc) : "memory");
}
DEVINL void wait_asynccnt0() {
  asm volatile("s_wait_asynccnt 0x0" ::: "memory");
}

// ------------------------------- converters --------------------------------

__global__ void f32_to_bf16_kernel(const float* __restrict__ src,
                                   bf16* __restrict__ dst, int n) {
  int i = blockIdx.x * blockDim.x + threadIdx.x;
  if (i < n) dst[i] = (bf16)src[i];
}

// Wt[n][k] = (bf16) W[k][n];  W is [Kdim][Ndim] row-major.
__global__ void transpose_f32_to_bf16_kernel(const float* __restrict__ W,
                                             bf16* __restrict__ Wt,
                                             int Kdim, int Ndim) {
  int i = blockIdx.x * blockDim.x + threadIdx.x;
  if (i < Kdim * Ndim) {
    int n = i / Kdim;
    int k = i - n * Kdim;
    Wt[i] = (bf16)W[k * Ndim + n];
  }
}

// ------------------------------- GEMM --------------------------------------
// C[M][N] = A[M][K] * Bt[N][K]^T + bias.  One wave -> 64x64 output tile
// (4x4 WMMA frags, 16 WMMAs per 32-deep k-step), k-loop unrolled x2 with
// double-buffered fragments so loads for step k+1 overlap WMMAs for step k.
// A-frag (16x32 bf16): lane holds row m=lane&15; K = 16*half + e (contiguous).
// B-frag (32x16 bf16): lane holds col n=lane&15; K = 16*half + e (contiguous).
// C-frag (16x16 f32):  element (m = i + 8*half, n = lane&15).

template <bool OUT_F32>
__global__ void __launch_bounds__(128)
gemm_bf16_kernel(const bf16* __restrict__ A,    // [Mdim][Kdim]
                 const bf16* __restrict__ Bt,   // [Ndim][Kdim]
                 const float* __restrict__ bias,// [Ndim]
                 void* __restrict__ Cout,
                 int Mdim, int Ndim, int Kdim) {
  const int lane   = threadIdx.x & 31;
  const int waveId = blockIdx.x * (blockDim.x >> 5) + (threadIdx.x >> 5);
  const int tilesN = Ndim >> 6;
  const int tm = waveId / tilesN;
  const int tn = waveId - tm * tilesN;
  const int m0 = tm * 64, n0 = tn * 64;
  if (m0 >= Mdim) return;

  const int nl   = lane & 15;
  const int half = lane >> 4;

  const bf16* ap[4];
  const bf16* bp[4];
#pragma unroll
  for (int u = 0; u < 4; ++u) {
    ap[u] = A  + (size_t)(m0 + 16 * u + nl) * Kdim + 16 * half;  // + kb, +8
    bp[u] = Bt + (size_t)(n0 + 16 * u + nl) * Kdim + 16 * half;  // + kb, +8
  }

  v8f acc[4][4] = {};
  v16bf a0[4], b0[4], a1[4], b1[4];

  // preload k-step 0
#pragma unroll
  for (int u = 0; u < 4; ++u) {
    a0[u] = cat16(*(const v8bf*)(ap[u]), *(const v8bf*)(ap[u] + 8));
    b0[u] = cat16(*(const v8bf*)(bp[u]), *(const v8bf*)(bp[u] + 8));
  }

  for (int kb = 0; kb < Kdim; kb += 64) {   // Kdim % 64 == 0
    // load k-step kb+32 while computing kb
#pragma unroll
    for (int u = 0; u < 4; ++u) {
      a1[u] = cat16(*(const v8bf*)(ap[u] + kb + 32), *(const v8bf*)(ap[u] + kb + 40));
      b1[u] = cat16(*(const v8bf*)(bp[u] + kb + 32), *(const v8bf*)(bp[u] + kb + 40));
    }
#pragma unroll
    for (int mu = 0; mu < 4; ++mu)
#pragma unroll
      for (int nu = 0; nu < 4; ++nu)
        acc[mu][nu] = wmma_bf16(a0[mu], b0[nu], acc[mu][nu]);

    // load k-step kb+64 while computing kb+32
    if (kb + 64 < Kdim) {
#pragma unroll
      for (int u = 0; u < 4; ++u) {
        a0[u] = cat16(*(const v8bf*)(ap[u] + kb + 64), *(const v8bf*)(ap[u] + kb + 72));
        b0[u] = cat16(*(const v8bf*)(bp[u] + kb + 64), *(const v8bf*)(bp[u] + kb + 72));
      }
    }
#pragma unroll
    for (int mu = 0; mu < 4; ++mu)
#pragma unroll
      for (int nu = 0; nu < 4; ++nu)
        acc[mu][nu] = wmma_bf16(a1[mu], b1[nu], acc[mu][nu]);
  }

#pragma unroll
  for (int nu = 0; nu < 4; ++nu) {
    const int c  = n0 + 16 * nu + nl;
    const float bb = bias[c];
#pragma unroll
    for (int mu = 0; mu < 4; ++mu) {
#pragma unroll
      for (int i = 0; i < 8; ++i) {
        const int rr = m0 + 16 * mu + i + 8 * half;
        if (OUT_F32)
          ((float*)Cout)[(size_t)rr * Ndim + c] = acc[mu][nu][i] + bb;
        else
          ((bf16*)Cout)[(size_t)rr * Ndim + c] = (bf16)(acc[mu][nu][i] + bb);
      }
    }
  }
}

// ------------------------------ attention ----------------------------------
// One wave per (batch, group, head-in-group, 16-query block).
// Flash-attention with causal mask over key tiles of 32.  The V tile is
// DMA'd into LDS asynchronously so it overlaps the S=QK^T WMMAs + softmax.

__global__ void __launch_bounds__(32)
attn_kernel(const bf16* __restrict__ Q,   // [M][E]   (head-major cols)
            const bf16* __restrict__ Km,  // [M][KV]
            const bf16* __restrict__ Vm,  // [M][KV]
            bf16* __restrict__ O) {       // [M][E]
  __shared__ bf16 Plds[16 * 32];     // P tile (16 q rows x 32 keys)
  __shared__ bf16 Vlds[32 * 128];    // V tile (32 keys x 128 dims)

  const int lane = threadIdx.x & 31;
  const int nl   = lane & 15;
  const int half = lane >> 4;

  int id = blockIdx.x;
  const int qt = id & 127; id >>= 7;     // T/16 = 128 query tiles
  const int r  = id & 3;   id >>= 2;
  const int g  = id & 3;   id >>= 2;
  const int bi = id;
  const int q0 = qt * 16;
  const int h  = g * R + r;

  // Q A-fragments for the whole 16x128 block (4 K-steps of 32).
  const bf16* qrow = Q + (size_t)(bi * T + q0 + nl) * E + h * HD;
  v16bf aq[4];
#pragma unroll
  for (int ks = 0; ks < 4; ++ks)
    aq[ks] = cat16(*(const v8bf*)(qrow + 32 * ks + 16 * half),
                   *(const v8bf*)(qrow + 32 * ks + 16 * half + 8));

  v8f o[8] = {};
  float mrow[8], lrow[8];
#pragma unroll
  for (int i = 0; i < 8; ++i) { mrow[i] = -INFINITY; lrow[i] = 0.0f; }

  const float scale = 0.08838834764831845f;   // 1/sqrt(128)
  const int ktiles = (q0 + 47) >> 5;          // causal bound

  for (int kt = 0; kt < ktiles; ++kt) {
    const int kk = kt * 32;

    // ---- kick off async DMA of this V tile into LDS (overlaps S + softmax)
    const bf16* vrow = Vm + (size_t)(bi * T + kk + lane) * KV + g * HD;
#pragma unroll
    for (int j = 0; j < 8; ++j)
      async_copy_b128(vrow + 8 * j, &Vlds[lane * 128 + 8 * j]);

    // ---- prefetch next key tile while we compute this one
    if (kt + 1 < ktiles) {
      __builtin_prefetch(Km + (size_t)(bi * T + kk + 32 + lane) * KV + g * HD, 0, 3);
      __builtin_prefetch(Vm + (size_t)(bi * T + kk + 32 + lane) * KV + g * HD, 0, 3);
    }

    // ---- S = Q * K^T: load all 8 B-frags first, then chain 8 WMMAs ----
    const bf16* k0 = Km + (size_t)(bi * T + kk + nl) * KV + g * HD + 16 * half;
    const bf16* k1 = Km + (size_t)(bi * T + kk + 16 + nl) * KV + g * HD + 16 * half;
    v16bf kf0[4], kf1[4];
#pragma unroll
    for (int ks = 0; ks < 4; ++ks) {
      kf0[ks] = cat16(*(const v8bf*)(k0 + 32 * ks), *(const v8bf*)(k0 + 32 * ks + 8));
      kf1[ks] = cat16(*(const v8bf*)(k1 + 32 * ks), *(const v8bf*)(k1 + 32 * ks + 8));
    }
    v8f s0 = {}, s1 = {};
#pragma unroll
    for (int ks = 0; ks < 4; ++ks) s0 = wmma_bf16(aq[ks], kf0[ks], s0);
#pragma unroll
    for (int ks = 0; ks < 4; ++ks) s1 = wmma_bf16(aq[ks], kf1[ks], s1);

    // ---- causal mask + online softmax update ----
    float p0[8], p1[8], alpha[8];
#pragma unroll
    for (int i = 0; i < 8; ++i) {
      const int qpos = q0 + i + 8 * half;
      float v0 = (kk + nl      <= qpos) ? s0[i] * scale : -INFINITY;
      float v1 = (kk + 16 + nl <= qpos) ? s1[i] * scale : -INFINITY;
      float mt = fmaxf(v0, v1);
#pragma unroll
      for (int msk = 8; msk >= 1; msk >>= 1)
        mt = fmaxf(mt, __shfl_xor(mt, msk, 32));
      const float mn = fmaxf(mrow[i], mt);
      const float a  = __expf(mrow[i] - mn);
      const float e0 = __expf(v0 - mn);
      const float e1 = __expf(v1 - mn);
      float sum = e0 + e1;
#pragma unroll
      for (int msk = 8; msk >= 1; msk >>= 1)
        sum += __shfl_xor(sum, msk, 32);
      lrow[i]  = lrow[i] * a + sum;
      mrow[i]  = mn;
      alpha[i] = a;
      p0[i] = e0;
      p1[i] = e1;
    }

    // rescale running output
#pragma unroll
    for (int j = 0; j < 8; ++j)
#pragma unroll
      for (int i = 0; i < 8; ++i)
        o[j][i] *= alpha[i];

    // ---- P (C-layout) -> LDS so it can be re-read in A-layout ----
#pragma unroll
    for (int i = 0; i < 8; ++i) {
      const int m = i + 8 * half;
      Plds[m * 32 + nl]      = (bf16)p0[i];
      Plds[m * 32 + 16 + nl] = (bf16)p1[i];
    }

    wait_asynccnt0();   // V tile DMA done
    __syncthreads();    // single-wave workgroup: dscnt drain + S_NOP barrier

    // P A-fragment: row m = nl, K = 16*half + e (contiguous)
    v16bf pa = cat16(*(const v8bf*)(&Plds[nl * 32 + 16 * half]),
                     *(const v8bf*)(&Plds[nl * 32 + 16 * half + 8]));

    // ---- O += P * V (8 column tiles of 16) ----
#pragma unroll
    for (int j = 0; j < 8; ++j) {
      v16bf vb;
#pragma unroll
      for (int e = 0; e < 16; ++e)
        vb[e] = Vlds[(16 * half + e) * 128 + 16 * j + nl];
      o[j] = wmma_bf16(pa, vb, o[j]);
    }
    __syncthreads();    // drain ds loads before next tile overwrites LDS
  }

  // ---- epilogue: divide by l, write bf16 attention output ----
#pragma unroll
  for (int j = 0; j < 8; ++j)
#pragma unroll
    for (int i = 0; i < 8; ++i) {
      const int m = i + 8 * half;
      O[(size_t)(bi * T + q0 + m) * E + h * HD + 16 * j + nl] =
          (bf16)(o[j][i] / lrow[i]);
    }
}

// ------------------------------- launcher ----------------------------------

extern "C" void kernel_launch(void* const* d_in, const int* in_sizes, int n_in,
                              void* d_out, int out_size, void* d_ws, size_t ws_size,
                              hipStream_t stream) {
  (void)in_sizes; (void)n_in; (void)out_size; (void)ws_size;

  const float* x  = (const float*)d_in[0];
  const float* Wq = (const float*)d_in[1];
  const float* bq = (const float*)d_in[2];
  const float* Wk = (const float*)d_in[3];
  const float* bk = (const float*)d_in[4];
  const float* Wv = (const float*)d_in[5];
  const float* bv = (const float*)d_in[6];
  const float* Wo = (const float*)d_in[7];
  const float* bo = (const float*)d_in[8];

  char* ws = (char*)d_ws;
  size_t off = 0;
  auto take = [&](size_t bytes) -> char* {
    char* p = ws + off;
    off += (bytes + 255) & ~(size_t)255;
    return p;
  };

  bf16* xb  = (bf16*)take((size_t)M * E * 2);
  bf16* WqT = (bf16*)take((size_t)E * E * 2);
  bf16* WkT = (bf16*)take((size_t)KV * E * 2);
  bf16* WvT = (bf16*)take((size_t)KV * E * 2);
  bf16* WoT = (bf16*)take((size_t)E * E * 2);
  bf16* Qb  = (bf16*)take((size_t)M * E * 2);
  bf16* Kb  = (bf16*)take((size_t)M * KV * 2);
  bf16* Vb  = (bf16*)take((size_t)M * KV * 2);
  bf16* Ab  = (bf16*)take((size_t)M * E * 2);

  // 1) precision conversion (+ weight transpose for WMMA B-fragments)
  {
    int n = M * E;
    f32_to_bf16_kernel<<<(n + 255) / 256, 256, 0, stream>>>(x, xb, n);
  }
  transpose_f32_to_bf16_kernel<<<(E * E + 255) / 256, 256, 0, stream>>>(Wq, WqT, E, E);
  transpose_f32_to_bf16_kernel<<<(E * KV + 255) / 256, 256, 0, stream>>>(Wk, WkT, E, KV);
  transpose_f32_to_bf16_kernel<<<(E * KV + 255) / 256, 256, 0, stream>>>(Wv, WvT, E, KV);
  transpose_f32_to_bf16_kernel<<<(E * E + 255) / 256, 256, 0, stream>>>(Wo, WoT, E, E);

  // 2) Q/K/V projections (bf16 WMMA, bf16 out), 64x64 tile per wave
  {
    int waves = (M / 64) * (E / 64);          // 2048
    gemm_bf16_kernel<false><<<waves / 4, 128, 0, stream>>>(xb, WqT, bq, Qb, M, E, E);
  }
  {
    int waves = (M / 64) * (KV / 64);         // 512
    gemm_bf16_kernel<false><<<waves / 4, 128, 0, stream>>>(xb, WkT, bk, Kb, M, KV, E);
    gemm_bf16_kernel<false><<<waves / 4, 128, 0, stream>>>(xb, WvT, bv, Vb, M, KV, E);
  }

  // 3) causal GQA flash attention (one wave per 16-query block per head)
  attn_kernel<<<BATCH * G * R * (T / 16), 32, 0, stream>>>(Qb, Kb, Vb, Ab);

  // 4) output projection (f32 out to d_out)
  {
    int waves = (M / 64) * (E / 64);
    gemm_bf16_kernel<true><<<waves / 4, 128, 0, stream>>>(Ab, WoT, bo, (float*)d_out, M, E, E);
  }
}
